// HGraphConvolutionNet_3401614098571
// MI455X (gfx1250) — compile-verified, matrix operand order checked by reading, not run on hardware
//
#include <hip/hip_runtime.h>
#include <math.h>

// ---------------------------------------------------------------------------
// HGraphConvolutionNet forward for MI455X (gfx1250, wave32, WMMA + TDM).
//
// Memory-bound on the 8192x8192 adjacency -> bf16 adj_norm (128MB), read
// exactly once per layer (block tile covers full N). All GEMMs use
// v_wmma_f32_16x16x32_bf16 (f32 accumulation).
//
// B operands are stored COLUMN-major (BT[n][k]) so both A and B fragments are
// two contiguous 8x bf16 runs per lane -> pure b128 traffic.
// B tiles are staged into LDS by the Tensor Data Mover (tensor_load_to_lds,
// double-buffered, TENSORcnt-tracked, issued by wave 0 only), overlapping the
// DMA with the WMMA stream; one workgroup barrier per 32-wide k-step.
// B fragments are software-pipelined one 16-col tile ahead so ds_load latency
// hides behind the previous WMMA instead of a full s_wait_dscnt 0 drain.
//
// Input pointer order assumption (setup_inputs dict insertion order, params
// flattened gcn-then-att, module-major, layer-minor, tuple order):
//   0:hs0 1:hs1 2:adj
//   3..14:  gcn (W,b) for m0L0,m0L1,m0L2,m1L0,m1L1,m1L2
//   15..32: att (Wh,bh,a) for m0L0..m0L2, m1L0..m1L2
//   33:splits0 34:splits1   (contiguous ranges -> block slicing)
// ---------------------------------------------------------------------------

#define NPER 4096
#define NTOT 8192

typedef __attribute__((ext_vector_type(16))) __bf16 v16bf;
typedef __attribute__((ext_vector_type(8)))  __bf16 v8bf;
typedef __attribute__((ext_vector_type(8)))  float  v8f;
typedef __attribute__((ext_vector_type(4)))  unsigned int v4u;
typedef __attribute__((ext_vector_type(8)))  int    v8i;
typedef __attribute__((ext_vector_type(4)))  int    v4i;

__device__ __forceinline__ __bf16 f2bf(float f) {
    unsigned u = __float_as_uint(f);
    unsigned r = (u + 0x7FFFu + ((u >> 16) & 1u)) >> 16;   // round-nearest-even
    return __builtin_bit_cast(__bf16, (unsigned short)r);
}

// --- degree: dinv[i] = rsqrt(sum_j adj[i][j] + 1)  (A = adj + I) -----------
__global__ __launch_bounds__(256)
void rowsum_rsqrt_kernel(const float* __restrict__ adj, float* __restrict__ dinv)
{
    __shared__ float red[256];
    const int row = blockIdx.x;
    float s = 0.f;
    for (int c = threadIdx.x; c < NTOT; c += 256)
        s += adj[(size_t)row * NTOT + c];
    red[threadIdx.x] = s;
    __syncthreads();
    for (int off = 128; off > 0; off >>= 1) {
        if (threadIdx.x < off) red[threadIdx.x] += red[threadIdx.x + off];
        __syncthreads();
    }
    if (threadIdx.x == 0) dinv[row] = rsqrtf(red[0] + 1.0f);
}

// --- adj_norm(bf16) = (adj + I) * dinv[i] * dinv[j] ------------------------
__global__ __launch_bounds__(256)
void normalize_adj_kernel(const float* __restrict__ adj,
                          const float* __restrict__ dinv,
                          __bf16* __restrict__ adjn)
{
    const int row = blockIdx.y;
    const int c0  = (blockIdx.x * 256 + threadIdx.x) * 4;
    const float dr = dinv[row];
#pragma unroll
    for (int i = 0; i < 4; ++i) {
        const int c = c0 + i;
        float v = adj[(size_t)row * NTOT + c] + ((c == row) ? 1.0f : 0.0f);
        adjn[(size_t)row * NTOT + c] = f2bf(v * dr * dinv[c]);
    }
}

// --- generic f32 -> bf16 ---------------------------------------------------
__global__ __launch_bounds__(256)
void f32_to_bf16_kernel(const float* __restrict__ src, __bf16* __restrict__ dst, int n)
{
    int i = blockIdx.x * 256 + threadIdx.x;
    if (i < n) dst[i] = f2bf(src[i]);
}

// --- W[in][out] f32  ->  WT[out][in] bf16 (GEMM-B column-major) ------------
__global__ __launch_bounds__(256)
void transpose_f32_to_bf16_kernel(const float* __restrict__ src,
                                  __bf16* __restrict__ dst, int in, int out)
{
    int idx = blockIdx.x * 256 + threadIdx.x;
    if (idx < in * out) {
        const int o = idx / in, i = idx - o * in;
        dst[idx] = f2bf(src[(size_t)i * out + o]);
    }
}

// --- batched bf16 WMMA GEMM with TDM-staged B ------------------------------
// C = A(MxK,row-major) @ BT(NxK,col-major-of-B)^T [+bias]
// Block: 256 thr = 8 waves, tile 128(M) x N (N = NT*16 <= 128). Wave w owns
// rows m0+w*16, keeps NT f32 accumulators, loads its A fragment straight from
// global (A has no reuse) and runs NT WMMAs per 32-wide k-step against the
// TDM-staged BT tile (double-buffered LDS, shared by all 8 waves).
struct GemmJob {
    const __bf16* A;      // [M][K] row-major
    const __bf16* BT;     // [N][K] (B column-major)
    float*   Cf;          // [M][ldc] f32 out (or null)
    __bf16*  CbT;         // [N][ldct] bf16 transposed out (or null)
    const float* bias;    // [N] (or null)
};
struct GemmBatch { GemmJob job[4]; };

template <int NT>
__global__ __launch_bounds__(256)
void gemm_bf16_wmma(GemmBatch batch, int lda, int ldbt, int ldc, int ldct, int K)
{
    const GemmJob jb = batch.job[blockIdx.y];
    constexpr int N = NT * 16;
    __shared__ __align__(16) __bf16 BsT[2][128 * 32];   // double-buffered B tile

    const int t    = threadIdx.x;
    const int lane = t & 31;
    const int w    = t >> 5;
    const int hl   = lane >> 4;
    const int ln16 = lane & 15;
    const int m0   = blockIdx.x * 128;

    // wave-uniform leader predicate (scalar branch; TDM ignores EXEC so we
    // must *branch* around it, not exec-mask it)
    const bool leader = (__builtin_amdgcn_readfirstlane(w) == 0);

    // Tensor DMA descriptor (CDNA5 ISA):
    // 2D tile: dim0 = 32 k (contiguous, 2B elems), dim1 = N rows, row stride
    // ldbt. group1 is k0-invariant; group0 carries lds/global addresses.
    v8i g1;
    g1[0] = 0x10000;                     // [17:16] data_size = 1 -> 2 bytes
    g1[1] = (int)(32u << 16);            // [63:48]  tensor_dim0 = 32
    g1[2] = (int)((unsigned)N << 16);    // [95:80]  tensor_dim1 = N
    g1[3] = (int)(32u << 16);            // [127:112] tile_dim0 = 32
    g1[4] = N;                           // [143:128] tile_dim1 = N
    g1[5] = ldbt;                        // [207:160] tensor_dim0_stride
    g1[6] = 0; g1[7] = 0;
    const v4i gz4 = {};
    const v8i gz8 = {};

    // LDS byte offsets (flat addr low 32 bits == LDS offset per ISA §10.2)
    const unsigned ldsB0 = (unsigned)(unsigned long long)(uintptr_t)&BsT[0][0];
    const unsigned ldsB1 = (unsigned)(unsigned long long)(uintptr_t)&BsT[1][0];
    const unsigned long long gBase = (unsigned long long)(uintptr_t)jb.BT;

    auto tdm_issue = [&](int k0, unsigned ldsAddr) {
        const unsigned long long ga = gBase + (unsigned long long)k0 * 2ull;
        v4u g0;
        g0[0] = 1u;                                   // count=1 (valid D#)
        g0[1] = ldsAddr;                              // lds_addr
        g0[2] = (unsigned)ga;                         // global_addr[31:0]
        g0[3] = (unsigned)((ga >> 32) & 0x1FFFFFFu)   // global_addr[56:32]
              | (2u << 30);                           // type = 2 ("image")
        // 6-arg form (clang-23 / therock-10.0 signature)
        __builtin_amdgcn_tensor_load_to_lds(g0, g1, gz4, gz4, gz8, 0);
    };

    if (leader) tdm_issue(0, ldsB0);                  // prime buffer 0

    v8f acc[NT];
#pragma unroll
    for (int nt = 0; nt < NT; ++nt) acc[nt] = (v8f){};

    // per-lane A fragment base: two contiguous runs at +0 / +16 each k-step
    const __bf16* aBase = jb.A + (size_t)(m0 + w * 16 + ln16) * lda + hl * 8;

    for (int k0 = 0, it = 0; k0 < K; k0 += 32, ++it) {
        const int cur = it & 1;
        if (leader) __builtin_amdgcn_s_wait_tensorcnt(0);  // cur buffer ready
        __syncthreads();                                    // publish to WG
        // overlap: DMA next tile into the other buffer while we compute; the
        // barrier above proved everyone finished reading it last iteration.
        if (leader && (k0 + 32 < K)) tdm_issue(k0 + 32, cur ? ldsB0 : ldsB1);

        const v8bf alo = *(const v8bf*)(aBase + k0);
        const v8bf ahi = *(const v8bf*)(aBase + k0 + 16);
        const v16bf fa = __builtin_shufflevector(alo, ahi,
            0, 1, 2, 3, 4, 5, 6, 7, 8, 9, 10, 11, 12, 13, 14, 15);
        if (k0 + 32 < K)
            __builtin_prefetch(aBase + k0 + 32, 0, 1);      // global_prefetch_b8

        // B fragments pipelined one tile ahead: loads for nt+1 are issued
        // before wmma(nt) so the ds latency hides behind the XDL op.
        const __bf16* Bcur = &BsT[cur][0];
        v8bf blo = *(const v8bf*)(Bcur + (size_t)ln16 * 32 + hl * 8);
        v8bf bhi = *(const v8bf*)(Bcur + (size_t)ln16 * 32 + hl * 8 + 16);
#pragma unroll
        for (int nt = 0; nt < NT; ++nt) {
            v8bf nlo = blo, nhi = bhi;
            if (nt + 1 < NT) {
                const __bf16* bp = Bcur + ((nt + 1) * 16 + ln16) * 32 + hl * 8;
                nlo = *(const v8bf*)(bp);
                nhi = *(const v8bf*)(bp + 16);
            }
            const v16bf fb = __builtin_shufflevector(blo, bhi,
                0, 1, 2, 3, 4, 5, 6, 7, 8, 9, 10, 11, 12, 13, 14, 15);
            acc[nt] = __builtin_amdgcn_wmma_f32_16x16x32_bf16(
                false, fa, false, fb, (short)0, acc[nt], false, false);
            blo = nlo; bhi = nhi;
        }
    }

    // epilogue: C/D layout -> col = lane&15, row = r + laneHalf*8
    const int rbase = m0 + w * 16 + hl * 8;
#pragma unroll
    for (int nt = 0; nt < NT; ++nt) {
        const int col = nt * 16 + ln16;
        const float bv = jb.bias ? jb.bias[col] : 0.0f;
        if (jb.Cf) {
            float* p = jb.Cf + (size_t)rbase * ldc + col;
#pragma unroll
            for (int r = 0; r < 8; ++r) p[(size_t)r * ldc] = acc[nt][r] + bv;
        }
        if (jb.CbT) {          // transposed bf16 out: 8 contiguous -> one b128
            v8bf o;
#pragma unroll
            for (int r = 0; r < 8; ++r) o[r] = f2bf(acc[nt][r] + bv);
            *(v8bf*)(jb.CbT + (size_t)col * ldct + rbase) = o;
        }
    }
}

// --- per-node attention combine (log-softmax weights, faithful to ref) -----
__global__ __launch_bounds__(128)
void attention_combine_kernel(const float* __restrict__ hpre_m1, // [2][4096][128]
                              const float* __restrict__ Wh,      // [F][50]
                              const float* __restrict__ bh,      // [50]
                              const float* __restrict__ av,      // [100]
                              int F, int m1, int do_relu,
                              float* __restrict__ hout,          // [8192][128]
                              __bf16* __restrict__ hcur)         // [8192][256] or null
{
    __shared__ float sh[2][50];
    __shared__ float lw[2];
    const int node = blockIdx.x;
    const int t = threadIdx.x;
    const float* h0 = hpre_m1 + (size_t)node * 128;
    const float* h1 = hpre_m1 + (size_t)NPER * 128 + (size_t)node * 128;

    const int mm = t >> 6, j = t & 63;
    if (j < 50) {
        const float* hm = mm ? h1 : h0;
        float s = bh[j];
        for (int f = 0; f < F; ++f) s += hm[f] * Wh[f * 50 + j];
        sh[mm][j] = s;
    }
    __syncthreads();
    if (t == 0) {
        float wv[2];
        for (int m = 0; m < 2; ++m) {
            float s = 0.f;
            for (int j2 = 0; j2 < 50; ++j2)
                s += sh[m][j2] * av[j2] + sh[m1][j2] * av[50 + j2];
            wv[m] = (s > 0.f) ? s : 0.01f * s;            // leaky_relu
        }
        float mx  = fmaxf(wv[0], wv[1]);
        float lse = mx + logf(expf(wv[0] - mx) + expf(wv[1] - mx));
        lw[0] = wv[0] - lse;                              // log_softmax, used as-is
        lw[1] = wv[1] - lse;
    }
    __syncthreads();
    if (t < F) {
        float v = lw[0] * h0[t] + lw[1] * h1[t];
        if (do_relu) v = fmaxf(v, 0.f);
        const size_t grow = (size_t)(m1 * NPER + node);
        hout[grow * 128 + t] = v;
        if (hcur) hcur[grow * 256 + t] = f2bf(v);
    }
}

// --- final: softmax(sigmoid(h), axis=1) over 16 classes --------------------
__global__ __launch_bounds__(256)
void final_softmax_kernel(const float* __restrict__ hout, float* __restrict__ out)
{
    const int node = blockIdx.x * 256 + threadIdx.x;
    if (node >= NTOT) return;
    float v[16], mx = -1e30f;
#pragma unroll
    for (int c = 0; c < 16; ++c) {
        float s = 1.f / (1.f + expf(-hout[(size_t)node * 128 + c]));
        v[c] = s; mx = fmaxf(mx, s);
    }
    float sum = 0.f;
#pragma unroll
    for (int c = 0; c < 16; ++c) { v[c] = expf(v[c] - mx); sum += v[c]; }
    const float inv = 1.f / sum;
#pragma unroll
    for (int c = 0; c < 16; ++c) out[(size_t)node * 16 + c] = v[c] * inv;
}

extern "C" void kernel_launch(void* const* d_in, const int* in_sizes, int n_in,
                              void* d_out, int out_size, void* d_ws, size_t ws_size,
                              hipStream_t stream)
{
    (void)in_sizes; (void)n_in; (void)out_size; (void)ws_size;

    const float* hs0 = (const float*)d_in[0];
    const float* hs1 = (const float*)d_in[1];
    const float* adj = (const float*)d_in[2];
    auto gcnW  = [&](int m, int L) { return (const float*)d_in[3  + m * 6 + L * 2]; };
    auto gcnB  = [&](int m, int L) { return (const float*)d_in[4  + m * 6 + L * 2]; };
    auto attWh = [&](int m, int L) { return (const float*)d_in[15 + m * 9 + L * 3]; };
    auto attBh = [&](int m, int L) { return (const float*)d_in[16 + m * 9 + L * 3]; };
    auto attA  = [&](int m, int L) { return (const float*)d_in[17 + m * 9 + L * 3]; };

    // workspace layout (~146 MiB)
    char* ws = (char*)d_ws;
    size_t off = 0;
    auto walloc = [&](size_t bytes) {
        void* p = ws + off; off = (off + bytes + 255) & ~(size_t)255; return p;
    };
    float*  dinv = (float*) walloc((size_t)NTOT * 4);
    __bf16* adjn = (__bf16*)walloc((size_t)NTOT * NTOT * 2);       // 128 MB
    __bf16* hcur = (__bf16*)walloc((size_t)NTOT * 256 * 2);        // activations, stride 256
    __bf16* wT   = (__bf16*)walloc((size_t)2 * 41984 * 2);         // gcn W^T bf16
    __bf16* hwT  = (__bf16*)walloc((size_t)2 * 128 * NPER * 2);    // (h@W)^T, [m][128][4096]
    float*  hpre = (float*) walloc((size_t)4 * NPER * 128 * 4);    // per (m1,m2) agg
    float*  hout = (float*) walloc((size_t)NTOT * 128 * 4);        // post-attention

    const int inD[3]   = {256, 128, 64};
    const int outD[3]  = {128, 64, 16};
    const int woffL[3] = {0, 256 * 128, 256 * 128 + 128 * 64};
    const int wmodStride = 256 * 128 + 128 * 64 + 64 * 16;         // 41984

    auto launchGemm = [&](const GemmBatch& b, int njobs, int lda, int ldbt,
                          int ldc, int ldct, int K, int NTtiles) {
        dim3 g(NPER / 128, njobs);
        switch (NTtiles) {
        case 8: gemm_bf16_wmma<8><<<g, 256, 0, stream>>>(b, lda, ldbt, ldc, ldct, K); break;
        case 4: gemm_bf16_wmma<4><<<g, 256, 0, stream>>>(b, lda, ldbt, ldc, ldct, K); break;
        default: gemm_bf16_wmma<1><<<g, 256, 0, stream>>>(b, lda, ldbt, ldc, ldct, K); break;
        }
    };

    // weights (transposed) + input features -> bf16
    for (int m = 0; m < 2; ++m)
        for (int L = 0; L < 3; ++L) {
            const int n = inD[L] * outD[L];
            transpose_f32_to_bf16_kernel<<<(n + 255) / 256, 256, 0, stream>>>(
                gcnW(m, L), wT + m * wmodStride + woffL[L], inD[L], outD[L]);
        }
    f32_to_bf16_kernel<<<(NPER * 256 + 255) / 256, 256, 0, stream>>>(hs0, hcur, NPER * 256);
    f32_to_bf16_kernel<<<(NPER * 256 + 255) / 256, 256, 0, stream>>>(
        hs1, hcur + (size_t)NPER * 256, NPER * 256);

    // adjacency normalization (once)
    rowsum_rsqrt_kernel<<<NTOT, 256, 0, stream>>>(adj, dinv);
    normalize_adj_kernel<<<dim3(NTOT / 1024, NTOT), 256, 0, stream>>>(adj, dinv, adjn);

    for (int L = 0; L < 3; ++L) {
        // HWT[m] = (hcur[m] @ W[m][L])^T    (bf16 transposed out)
        GemmBatch hb = {};
        for (int m = 0; m < 2; ++m) {
            hb.job[m].A    = hcur + (size_t)m * NPER * 256;
            hb.job[m].BT   = wT + m * wmodStride + woffL[L];
            hb.job[m].Cf   = nullptr;
            hb.job[m].CbT  = hwT + (size_t)m * 128 * NPER;
            hb.job[m].bias = nullptr;
        }
        launchGemm(hb, 2, /*lda*/256, /*ldbt*/inD[L], /*ldc*/0, /*ldct*/NPER,
                   /*K*/inD[L], outD[L] / 16);

        // hpre[m1][m2] = Ar(m1,m2) @ HW[m2] + b[m2][L]   (f32 out; BT = HWT)
        GemmBatch ab = {};
        for (int m1 = 0; m1 < 2; ++m1)
            for (int m2 = 0; m2 < 2; ++m2) {
                GemmJob& j = ab.job[m1 * 2 + m2];
                j.A    = adjn + (size_t)m1 * NPER * NTOT + (size_t)m2 * NPER;
                j.BT   = hwT + (size_t)m2 * 128 * NPER;
                j.Cf   = hpre + (size_t)(m1 * 2 + m2) * NPER * 128;
                j.CbT  = nullptr;
                j.bias = gcnB(m2, L);
            }
        launchGemm(ab, 4, /*lda*/NTOT, /*ldbt*/NPER, /*ldc*/128, /*ldct*/0,
                   /*K*/NPER, outD[L] / 16);

        // attention combine (+relu +bf16 re-quant for next layer)
        for (int m1 = 0; m1 < 2; ++m1)
            attention_combine_kernel<<<NPER, 128, 0, stream>>>(
                hpre + (size_t)m1 * 2 * NPER * 128,
                attWh(m1, L), attBh(m1, L), attA(m1, L),
                outD[L], m1, (L < 2) ? 1 : 0,
                hout, (L < 2) ? hcur : (__bf16*)nullptr);
    }

    final_softmax_kernel<<<NTOT / 256, 256, 0, stream>>>(hout, (float*)d_out);
}
